// Quantize_833223656390
// MI455X (gfx1250) — matrix operand order repaired and checked by reading, not compiled, not run on hardware
//
#include <hip/hip_runtime.h>
#include <hip/hip_bf16.h>
#include <stdint.h>

// ---------------------------------------------------------------------------
// VQ codebook quantize for MI455X (gfx1250).
//   x:     [8,2048,512] f32   -> N=16384 rows, D=512
//   embed: [512,8192]   f32   (columns L2-normalized)
// dist = ||xn||^2 - 2 xn.e + ||e||^2 = 2 - 2 xn.e  (both unit)  =>
// argmin dist == argmax dot  => fused GEMM(16384x8192x512) + row argmax.
// 137 GFLOP vs ~120MB HBM => compute bound => v_wmma_f32_16x16x32_bf16.
// Inner loop is LDS-bw bound, so each wave register-blocks 32x64 (8 C tiles):
// 12 ds_load_b128 feed 8 WMMAs per K-step (1.5 loads/WMMA vs 4 naive).
// ---------------------------------------------------------------------------

typedef __attribute__((ext_vector_type(8)))  __bf16 v8bf;
typedef __attribute__((ext_vector_type(16))) __bf16 v16bf;
typedef __attribute__((ext_vector_type(8)))  float  v8f;

#define N_ROWS 16384
#define DDIM   512
#define N_EMB  8192
#define AROWS  128          // rows of A staged per block
#define BCOLS  128          // cols of B staged per block (per iteration)
#define PITCH  520          // LDS row pitch in bf16 (512 + 8): +4 dword-banks/row
#define NSPLIT 8            // column splits (grid.y) for chip-wide occupancy
#define ITERS_PER_SPLIT (N_EMB / BCOLS / NSPLIT)   // 8 x 128-col iterations

__device__ __forceinline__ unsigned short f2bf(float f) {
  unsigned int u = __float_as_uint(f);                 // round-to-nearest-even
  return (unsigned short)((u + 0x7FFFu + ((u >> 16) & 1u)) >> 16);
}

// ---- embed [512][8192] f32  ->  embedT [8192][512] as bf16 + f32 -----------
__global__ void k_prep_embed(const float* __restrict__ embed,
                             unsigned short* __restrict__ eTbf,
                             float* __restrict__ eTf) {
  __shared__ float tile[32][33];
  int n0 = blockIdx.x * 32, k0 = blockIdx.y * 32;
  for (int i = threadIdx.y; i < 32; i += 8)
    tile[i][threadIdx.x] = embed[(size_t)(k0 + i) * N_EMB + n0 + threadIdx.x];
  __syncthreads();
  for (int i = threadIdx.y; i < 32; i += 8) {
    float v  = tile[threadIdx.x][i];
    size_t o = (size_t)(n0 + i) * DDIM + k0 + threadIdx.x;   // coalesced in k
    eTf[o]  = v;
    eTbf[o] = f2bf(v);
  }
}

// ---- row-normalize x -> bf16 ------------------------------------------------
__global__ void k_normalize(const float* __restrict__ x,
                            unsigned short* __restrict__ xnbf) {
  const int row = blockIdx.x, t = threadIdx.x;            // 128 threads
  const float4 v = *(const float4*)(x + (size_t)row * DDIM + t * 4);
  float ss = v.x * v.x + v.y * v.y + v.z * v.z + v.w * v.w;
  for (int m = 16; m >= 1; m >>= 1) ss += __shfl_xor(ss, m, 32);
  __shared__ float sred[4];
  if ((t & 31) == 0) sred[t >> 5] = ss;
  __syncthreads();
  float tot = sred[0] + sred[1] + sred[2] + sred[3];
  float inv = 1.0f / fmaxf(sqrtf(tot), 1e-12f);
  unsigned int lo = (unsigned int)f2bf(v.x * inv) | ((unsigned int)f2bf(v.y * inv) << 16);
  unsigned int hi = (unsigned int)f2bf(v.z * inv) | ((unsigned int)f2bf(v.w * inv) << 16);
  uint2 p; p.x = lo; p.y = hi;
  *(uint2*)(xnbf + (size_t)row * DDIM + t * 4) = p;
}

// ---- main: GEMM + fused row argmax (WMMA bf16, f32 accumulate) -------------
// Block: 256 threads = 8 waves, arranged 4 (row groups) x 2 (col groups).
// Wave computes 32 rows x 64 cols = 2x4 = 8 WMMA accumulator tiles.
__global__ __launch_bounds__(256)
void k_argmax(const unsigned short* __restrict__ xnbf,
              const unsigned short* __restrict__ eTbf,
              float* __restrict__ pbest, int* __restrict__ pidx) {
  extern __shared__ __align__(16) char smem[];
  unsigned short* As = (unsigned short*)smem;                       // 128 x 520
  unsigned short* Bs = (unsigned short*)(smem + AROWS * PITCH * 2); // 128 x 520

  const int tid  = threadIdx.x;
  const int wave = tid >> 5, lane = tid & 31;
  const int lm   = lane & 15, half = lane >> 4;
  const int rg   = wave >> 1;                 // row group 0..3  (32 rows each)
  const int cg   = wave & 1;                  // col group 0..1  (64 cols each)
  const int row0 = blockIdx.x * AROWS;
  const int colSplit0 = blockIdx.y * (N_EMB / NSPLIT);

  // Stage A panel: 128 rows x 512 bf16, 16B chunks, coalesced.
  for (int ch = tid; ch < AROWS * 64; ch += 256) {
    int r = ch >> 6, o = (ch & 63) << 3;
    *(uint4*)(As + r * PITCH + o) =
        *(const uint4*)(xnbf + (size_t)(row0 + r) * DDIM + o);
  }

  float best[16]; int bidx[16];               // [rt*8+i] for rows of this lane
#pragma unroll
  for (int i = 0; i < 16; ++i) { best[i] = -3.4028235e38f; bidx[i] = 0; }

  // Per-lane fragment base pointers (contiguous k = 16*half + s for A and B, not the ISA table order)
  const unsigned short* aB0 = As + (rg * 32 + lm) * PITCH + half * 16;       // rt=0
  const unsigned short* aB1 = aB0 + 16 * PITCH;                               // rt=1
  const unsigned short* bB0 = Bs + (cg * 64 + lm) * PITCH + half * 16;        // ct=0
  const unsigned short* bB1 = bB0 + 16 * PITCH;
  const unsigned short* bB2 = bB0 + 32 * PITCH;
  const unsigned short* bB3 = bB0 + 48 * PITCH;

  for (int it = 0; it < ITERS_PER_SPLIT; ++it) {
    const int colbase = colSplit0 + it * BCOLS;
    __syncthreads();                       // prev compute done before B overwrite
    // Stage B panel: 128 cols x 512 bf16 = contiguous 128KB region of embedT.
    for (int ch = tid; ch < BCOLS * 64; ch += 256) {
      int c = ch >> 6, o = (ch & 63) << 3;
      *(uint4*)(Bs + c * PITCH + o) =
          *(const uint4*)(eTbf + ((size_t)(colbase + c) << 9) + o);
    }
    __syncthreads();

    v8f acc[2][4] = {};
#pragma unroll
    for (int kt = 0; kt < 16; ++kt) {      // K = 512 = 16 x 32
      const int kb = kt << 5;
      // 2 A fragments (rows), 4 B fragments (cols)
      v8bf a0l = *(const v8bf*)(aB0 + kb), a0h = *(const v8bf*)(aB0 + kb + 8);
      v8bf a1l = *(const v8bf*)(aB1 + kb), a1h = *(const v8bf*)(aB1 + kb + 8);
      v16bf a0 = __builtin_shufflevector(a0l, a0h, 0,1,2,3,4,5,6,7,8,9,10,11,12,13,14,15);
      v16bf a1 = __builtin_shufflevector(a1l, a1h, 0,1,2,3,4,5,6,7,8,9,10,11,12,13,14,15);
      v8bf b0l = *(const v8bf*)(bB0 + kb), b0h = *(const v8bf*)(bB0 + kb + 8);
      v8bf b1l = *(const v8bf*)(bB1 + kb), b1h = *(const v8bf*)(bB1 + kb + 8);
      v8bf b2l = *(const v8bf*)(bB2 + kb), b2h = *(const v8bf*)(bB2 + kb + 8);
      v8bf b3l = *(const v8bf*)(bB3 + kb), b3h = *(const v8bf*)(bB3 + kb + 8);
      v16bf b0 = __builtin_shufflevector(b0l, b0h, 0,1,2,3,4,5,6,7,8,9,10,11,12,13,14,15);
      v16bf b1 = __builtin_shufflevector(b1l, b1h, 0,1,2,3,4,5,6,7,8,9,10,11,12,13,14,15);
      v16bf b2 = __builtin_shufflevector(b2l, b2h, 0,1,2,3,4,5,6,7,8,9,10,11,12,13,14,15);
      v16bf b3 = __builtin_shufflevector(b3l, b3h, 0,1,2,3,4,5,6,7,8,9,10,11,12,13,14,15);
      acc[0][0] = __builtin_amdgcn_wmma_f32_16x16x32_bf16(false, a0, false, b0, (short)0, acc[0][0], false, false);
      acc[0][1] = __builtin_amdgcn_wmma_f32_16x16x32_bf16(false, a0, false, b1, (short)0, acc[0][1], false, false);
      acc[0][2] = __builtin_amdgcn_wmma_f32_16x16x32_bf16(false, a0, false, b2, (short)0, acc[0][2], false, false);
      acc[0][3] = __builtin_amdgcn_wmma_f32_16x16x32_bf16(false, a0, false, b3, (short)0, acc[0][3], false, false);
      acc[1][0] = __builtin_amdgcn_wmma_f32_16x16x32_bf16(false, a1, false, b0, (short)0, acc[1][0], false, false);
      acc[1][1] = __builtin_amdgcn_wmma_f32_16x16x32_bf16(false, a1, false, b1, (short)0, acc[1][1], false, false);
      acc[1][2] = __builtin_amdgcn_wmma_f32_16x16x32_bf16(false, a1, false, b2, (short)0, acc[1][2], false, false);
      acc[1][3] = __builtin_amdgcn_wmma_f32_16x16x32_bf16(false, a1, false, b3, (short)0, acc[1][3], false, false);
    }
    // Running argmax; cols ascend (ct, then it) so strict > keeps lowest index
#pragma unroll
    for (int rt = 0; rt < 2; ++rt) {
#pragma unroll
      for (int ct = 0; ct < 4; ++ct) {
        const int col = colbase + cg * 64 + ct * 16 + lm;
#pragma unroll
        for (int i = 0; i < 8; ++i) {
          float v = acc[rt][ct][i];        // C vgpr i <-> row i+8*half, col lm
          if (v > best[rt * 8 + i]) { best[rt * 8 + i] = v; bidx[rt * 8 + i] = col; }
        }
      }
    }
  }

  // Reduce across the 16 column-lanes of this half (xor stays within half).
#pragma unroll
  for (int m = 1; m < 16; m <<= 1) {
#pragma unroll
    for (int i = 0; i < 16; ++i) {
      float ov = __shfl_xor(best[i], m, 32);
      int   oi = __shfl_xor(bidx[i], m, 32);
      if (ov > best[i] || (ov == best[i] && oi < bidx[i])) { best[i] = ov; bidx[i] = oi; }
    }
  }
  if (lm == 0) {
#pragma unroll
    for (int rt = 0; rt < 2; ++rt)
#pragma unroll
      for (int i = 0; i < 8; ++i) {
        int r = row0 + rg * 32 + rt * 16 + half * 8 + i;
        pbest[(size_t)blockIdx.y * N_ROWS + r] = best[rt * 8 + i];
        pidx [(size_t)blockIdx.y * N_ROWS + r] = bidx[rt * 8 + i];
      }
  }
}

// ---- merge the NSPLIT partial argmaxes -------------------------------------
__global__ void k_pick(const float* __restrict__ pbest, const int* __restrict__ pidx,
                       int* __restrict__ bestIdx) {
  int i = blockIdx.x * 256 + threadIdx.x;
  if (i >= N_ROWS) return;
  float bv = pbest[i]; int bi = pidx[i];
  for (int s = 1; s < NSPLIT; ++s) {
    float v = pbest[(size_t)s * N_ROWS + i];
    int   j = pidx [(size_t)s * N_ROWS + i];
    if (v > bv || (v == bv && j < bi)) { bv = v; bi = j; }
  }
  bestIdx[i] = bi;
}

// ---- zero count stats ------------------------------------------------------
__global__ void k_zero(float* __restrict__ counts) {
  int i = blockIdx.x * 256 + threadIdx.x;
  if (i < N_EMB) counts[i] = 0.0f;
}

// ---- gather quantize, diff partials, counts, indices -----------------------
__global__ void k_gather(const float* __restrict__ x, const float* __restrict__ eTf,
                         const int* __restrict__ bestIdx,
                         float* __restrict__ qOut, float* __restrict__ indOut,
                         float* __restrict__ counts, float* __restrict__ diffpart) {
  const int row = blockIdx.x, t = threadIdx.x;            // 128 threads
  const int j = bestIdx[row];
  const float4 xv = *(const float4*)(x + (size_t)row * DDIM + t * 4);
  float ss = xv.x * xv.x + xv.y * xv.y + xv.z * xv.z + xv.w * xv.w;
  for (int m = 16; m >= 1; m >>= 1) ss += __shfl_xor(ss, m, 32);
  __shared__ float sss[4]; __shared__ float sad[4];
  if ((t & 31) == 0) sss[t >> 5] = ss;
  __syncthreads();
  float inv = 1.0f / fmaxf(sqrtf(sss[0] + sss[1] + sss[2] + sss[3]), 1e-12f);

  const float4 q = *(const float4*)(eTf + (size_t)j * DDIM + t * 4);
  *(float4*)(qOut + (size_t)row * DDIM + t * 4) = q;      // STE fwd == quantize
  float ad = fabsf(q.x - xv.x * inv) + fabsf(q.y - xv.y * inv)
           + fabsf(q.z - xv.z * inv) + fabsf(q.w - xv.w * inv);
  for (int m = 16; m >= 1; m >>= 1) ad += __shfl_xor(ad, m, 32);
  if ((t & 31) == 0) sad[t >> 5] = ad;
  __syncthreads();
  if (t == 0) {
    diffpart[row] = sad[0] + sad[1] + sad[2] + sad[3];    // deterministic reduce later
    atomicAdd(&counts[j], 1.0f);                          // integer-valued: exact
    indOut[row] = (float)j;
  }
}

// ---- prob, perplexity, diff ------------------------------------------------
__global__ void k_stats(const float* __restrict__ counts, const float* __restrict__ diffpart,
                        float* __restrict__ probOut, float* __restrict__ perpOut,
                        float* __restrict__ diffOut) {
  __shared__ float red[256];
  const int t = threadIdx.x;
  float s = 0.0f;
  for (int i = t; i < N_EMB; i += 256) s += counts[i];
  red[t] = s; __syncthreads();
  for (int k = 128; k > 0; k >>= 1) { if (t < k) red[t] += red[t + k]; __syncthreads(); }
  const float denom = red[0] + 1e-5f;
  __syncthreads();
  float pl = 0.0f;
  for (int i = t; i < N_EMB; i += 256) {
    float p = counts[i] / denom;
    probOut[i] = p;
    pl += p * logf(fmaxf(p, 1e-5f));
  }
  red[t] = pl; __syncthreads();
  for (int k = 128; k > 0; k >>= 1) { if (t < k) red[t] += red[t + k]; __syncthreads(); }
  const float plogp = red[0];
  __syncthreads();
  float ds = 0.0f;
  for (int i = t; i < N_ROWS; i += 256) ds += diffpart[i];
  red[t] = ds; __syncthreads();
  for (int k = 128; k > 0; k >>= 1) { if (t < k) red[t] += red[t + k]; __syncthreads(); }
  if (t == 0) {
    perpOut[0] = expf(-plogp);
    diffOut[0] = red[0] / (float)((size_t)N_ROWS * DDIM);
  }
}

// ---------------------------------------------------------------------------
extern "C" void kernel_launch(void* const* d_in, const int* in_sizes, int n_in,
                              void* d_out, int out_size, void* d_ws, size_t ws_size,
                              hipStream_t stream) {
  (void)in_sizes; (void)n_in; (void)out_size; (void)ws_size;
  const float* x     = (const float*)d_in[0];   // [16384,512]
  const float* embed = (const float*)d_in[1];   // [512,8192]

  // workspace layout (bytes)
  char* ws = (char*)d_ws;
  unsigned short* xnbf = (unsigned short*)ws;                    // 16,777,216
  unsigned short* eTbf = (unsigned short*)(ws + 16777216);       //  8,388,608
  float*  eTf      = (float*)(ws + 25165824);                    // 16,777,216
  int*    bestIdx  = (int*)  (ws + 41943040);                    //     65,536
  float*  counts   = (float*)(ws + 42008576);                    //     32,768
  float*  diffpart = (float*)(ws + 42041344);                    //     65,536
  float*  pbest    = (float*)(ws + 42106880);                    //    524,288
  int*    pidx     = (int*)  (ws + 42631168);                    //    524,288

  // output layout (floats): quantize | diff | embed_ind | prob | perplexity
  float* out     = (float*)d_out;
  float* qOut    = out;
  float* diffOut = out + 8388608;
  float* indOut  = out + 8388609;
  float* probOut = out + 8404993;
  float* perpOut = out + 8413185;

  k_zero      <<<(N_EMB + 255) / 256, 256, 0, stream>>>(counts);
  k_prep_embed<<<dim3(N_EMB / 32, DDIM / 32), dim3(32, 8), 0, stream>>>(embed, eTbf, eTf);
  k_normalize <<<N_ROWS, 128, 0, stream>>>(x, xnbf);

  const size_t smem = (size_t)(AROWS + BCOLS) * PITCH * 2;   // 266,240 B (<320KB/WG)
  k_argmax<<<dim3(N_ROWS / AROWS, NSPLIT), 256, smem, stream>>>(xnbf, eTbf, pbest, pidx);

  k_pick  <<<(N_ROWS + 255) / 256, 256, 0, stream>>>(pbest, pidx, bestIdx);
  k_gather<<<N_ROWS, 128, 0, stream>>>(x, eTf, bestIdx, qOut, indOut, counts, diffpart);
  k_stats <<<1, 256, 0, stream>>>(counts, diffpart, probOut, perpOut, diffOut);
}
